// MultiHeadAttention_87016037417034
// MI455X (gfx1250) — matrix operand order repaired and checked by reading, not compiled, not run on hardware
//
#include <hip/hip_runtime.h>
#include <hip/hip_bf16.h>

// ---------------------------------------------------------------------------
// MHA forward for MI455X (gfx1250, wave32, WMMA).
//   y = softmax((XqWq^T)(XkWk^T)^T) (XvWv^T) Wo^T + biases
// All GEMMs + attention use v_wmma_f32_16x16x32_bf16 with f32 accumulation.
// Flash-attention streaming softmax: scores never touch HBM.
// GEMM LDS staging uses CDNA5 async global->LDS copies when available.
// ---------------------------------------------------------------------------

typedef __attribute__((ext_vector_type(16))) __bf16         v16bf;
typedef __attribute__((ext_vector_type(8)))  float          v8f;
typedef __attribute__((ext_vector_type(4)))  unsigned int   uint4v;
typedef __attribute__((ext_vector_type(4)))  int            int4v;
typedef __attribute__((ext_vector_type(4)))  float          float4v;
typedef __attribute__((ext_vector_type(4)))  unsigned short ushort4v;

typedef __attribute__((address_space(1))) int4v g_int4;   // global int4
typedef __attribute__((address_space(3))) int4v l_int4;   // LDS int4

constexpr int S_ = 2048, B_ = 2, D_ = 1024, H_ = 16, DK_ = 64;
constexpr int M_ = S_ * B_;     // 4096 rows for all projections

#if __has_builtin(__builtin_amdgcn_global_load_async_to_lds_b128) && \
    __has_builtin(__builtin_amdgcn_s_wait_asynccnt)
#define USE_ASYNC_LDS 1
#else
#define USE_ASYNC_LDS 0
#endif

// Per-lane 16B global -> LDS copy (ASYNCcnt path on CDNA5).
__device__ __forceinline__ void copy16B_g2l(const unsigned short* g,
                                            unsigned short* l) {
#if USE_ASYNC_LDS
  __builtin_amdgcn_global_load_async_to_lds_b128((g_int4*)g, (l_int4*)l, 0, 0);
#else
  *(uint4v*)l = *(const uint4v*)g;
#endif
}
__device__ __forceinline__ void copy_g2l_join() {
#if USE_ASYNC_LDS
  __builtin_amdgcn_s_wait_asynccnt(0);
#endif
}

// float -> bf16, round to nearest even
__device__ __forceinline__ unsigned short f2bf(float f) {
  unsigned int u = __builtin_bit_cast(unsigned int, f);
  u += 0x7FFFu + ((u >> 16) & 1u);
  return (unsigned short)(u >> 16);
}

__device__ __forceinline__ v8f wmma_bf16(v16bf a, v16bf b, v8f c) {
  // D = A(16x32) * B(32x16) + C(16x16), f32 accumulate
  return __builtin_amdgcn_wmma_f32_16x16x32_bf16(
      /*neg_a=*/false, a, /*neg_b=*/false, b,
      /*c_mod=*/(short)0, c, /*reuse_a=*/false, /*reuse_b=*/false);
}

// B-fragment load: 16 contiguous bf16 (column n = lane%16, K half by lane/16).
__device__ __forceinline__ v16bf load_frag_k16(const unsigned short* p) {
  union { uint4v u[2]; v16bf v; } x;
  x.u[0] = *(const uint4v*)(p);
  x.u[1] = *(const uint4v*)(p + 8);
  return x.v;
}
// A-fragment load: 16 contiguous bf16 at p (row m = lane%16, K half by lane/16).
__device__ __forceinline__ v16bf load_frag_a(const unsigned short* p) {
  union { uint4v u[2]; v16bf v; } x;
  x.u[0] = *(const uint4v*)(p);
  x.u[1] = *(const uint4v*)(p + 8);
  return x.v;
}

// ---------------------------------------------------------------------------
// Kernel 1: f32 -> bf16 conversion (vector x4)
// ---------------------------------------------------------------------------
__global__ __launch_bounds__(256) void cvt_bf16_kernel(
    const float4v* __restrict__ in, ushort4v* __restrict__ out, int n4) {
  int i = blockIdx.x * blockDim.x + threadIdx.x;
  if (i < n4) {
    float4v f = in[i];
    ushort4v r;
    r.x = f2bf(f.x); r.y = f2bf(f.y); r.z = f2bf(f.z); r.w = f2bf(f.w);
    out[i] = r;
  }
}

// ---------------------------------------------------------------------------
// Kernel 2: bf16 GEMM  C[M,N] = A[M,K] * W[N,K]^T + bias[N]
// Block: 128x128 tile of C, 256 threads = 8 waves, each wave 64x32.
// K-step 64 staged through LDS (async copies, padded stride 72 halfwords).
// MODE 0: bf16 row-major [M,N]   (Q, K projections)
// MODE 1: bf16 transposed per head -> [B,H,DK,S]  (V projection)
// MODE 2: f32 row-major [M,N]    (output projection)
// ---------------------------------------------------------------------------
template <int MODE>
__global__ __launch_bounds__(256) void gemm_bf16_kernel(
    const unsigned short* __restrict__ A,   // [M,K] bf16 row-major
    const unsigned short* __restrict__ W,   // [N,K] bf16 row-major
    const float* __restrict__ bias,         // [N] f32
    void* __restrict__ out, int Mdim, int Ndim, int Kdim) {
  constexpr int BM = 128, BN = 128, BK = 64, LDT = BK + 8;  // pad: no conflicts
  __shared__ unsigned short As[BM * LDT];   // 18 KiB
  __shared__ unsigned short Bs[BN * LDT];   // 18 KiB

  const int t    = threadIdx.x;
  const int lane = t & 31;
  const int ln   = lane & 15;      // column / row within 16
  const int lh   = lane >> 4;      // lane half
  const int w    = t >> 5;
  const int wm   = (w & 1) * 64;   // wave M offset in block
  const int wn   = (w >> 1) * 32;  // wave N offset in block
  const int bm   = blockIdx.y * BM;
  const int bn   = blockIdx.x * BN;

  v8f acc[4][2];
  const v8f vzero = {0.f, 0.f, 0.f, 0.f, 0.f, 0.f, 0.f, 0.f};
#pragma unroll
  for (int mt = 0; mt < 4; ++mt)
#pragma unroll
    for (int nt = 0; nt < 2; ++nt) acc[mt][nt] = vzero;

  const int lrow = t & 127;         // 0..127
  const int lseg = (t >> 7) * 32;   // 0 or 32 (halfwords): each thread 64B/row

  for (int k0 = 0; k0 < Kdim; k0 += BK) {
    __syncthreads();
    const unsigned short* ag = A + (size_t)(bm + lrow) * Kdim + k0 + lseg;
    const unsigned short* bg = W + (size_t)(bn + lrow) * Kdim + k0 + lseg;
    unsigned short* as = As + lrow * LDT + lseg;
    unsigned short* bs = Bs + lrow * LDT + lseg;
    __builtin_prefetch(ag + BK, 0, 1);   // global_prefetch next K tile
    __builtin_prefetch(bg + BK, 0, 1);
#pragma unroll
    for (int c = 0; c < 4; ++c) copy16B_g2l(ag + c * 8, as + c * 8);
#pragma unroll
    for (int c = 0; c < 4; ++c) copy16B_g2l(bg + c * 8, bs + c * 8);
    copy_g2l_join();                     // s_wait_asynccnt 0
    __syncthreads();

#pragma unroll
    for (int ks = 0; ks < BK; ks += 32) {
      v16bf bfrag[2];
#pragma unroll
      for (int nt = 0; nt < 2; ++nt)
        bfrag[nt] =
            load_frag_k16(Bs + (wn + nt * 16 + ln) * LDT + ks + lh * 16);
#pragma unroll
      for (int mt = 0; mt < 4; ++mt) {
        v16bf afrag = load_frag_a(As + (wm + mt * 16 + ln) * LDT + ks + lh * 16);
#pragma unroll
        for (int nt = 0; nt < 2; ++nt)
          acc[mt][nt] = wmma_bf16(afrag, bfrag[nt], acc[mt][nt]);
      }
    }
  }

  // Epilogue: bias + store per MODE.
#pragma unroll
  for (int mt = 0; mt < 4; ++mt) {
#pragma unroll
    for (int nt = 0; nt < 2; ++nt) {
#pragma unroll
      for (int v = 0; v < 8; ++v) {
        const int gm = bm + wm + mt * 16 + v + (lh << 3);  // row (s*B + b)
        const int gn = bn + wn + nt * 16 + ln;             // col (h*DK + dk)
        const float val = acc[mt][nt][v] + bias[gn];
        if (MODE == 0) {
          ((unsigned short*)out)[(size_t)gm * Ndim + gn] = f2bf(val);
        } else if (MODE == 1) {
          const int s = gm / B_, bb = gm % B_;
          const int h = gn / DK_, dk = gn % DK_;
          ((unsigned short*)out)[(((size_t)bb * H_ + h) * DK_ + dk) * S_ + s] =
              f2bf(val);
        } else {
          ((float*)out)[(size_t)gm * Ndim + gn] = val;
        }
      }
    }
  }
}

// ---------------------------------------------------------------------------
// Kernel 3: flash attention, unscaled QK^T (faithful to reference).
// Grid: (S/64, B*H). Block: 128 threads = 4 independent waves.
// Each wave: 16 query rows, streams keys 64 at a time (16 WMMA / iter).
//   Q,K bf16 [S*B, D]; V bf16 transposed [B,H,DK,S]; out AO bf16 [S*B, D].
// ---------------------------------------------------------------------------
__global__ __launch_bounds__(128) void flash_attn_kernel(
    const unsigned short* __restrict__ Qb, const unsigned short* __restrict__ Kb,
    const unsigned short* __restrict__ Vt, unsigned short* __restrict__ AO) {
  constexpr int PLD = 72;                       // padded LDS row stride
  __shared__ unsigned short Plds[4][16 * PLD];  // per-wave P tile (16x64 bf16)

  const int t = threadIdx.x, lane = t & 31, w = t >> 5;
  const int ln = lane & 15, lh = lane >> 4;
  const int bh = blockIdx.y, b = bh / H_, h = bh % H_;
  const int q0 = blockIdx.x * 64 + w * 16;
  unsigned short* Pl = &Plds[w][0];

  // Preload Q A-fragments (DK=64 -> 2 fragments of 16x32).
  v16bf aq[2];
  {
    const int srow = q0 + ln;
    const unsigned short* qp = Qb + ((size_t)srow * B_ + b) * D_ + h * DK_;
#pragma unroll
    for (int kq = 0; kq < 2; ++kq) aq[kq] = load_frag_a(qp + kq * 32 + lh * 16);
  }

  float m8[8], l8[8];
  v8f o[4];
  const v8f vzero = {0.f, 0.f, 0.f, 0.f, 0.f, 0.f, 0.f, 0.f};
#pragma unroll
  for (int v = 0; v < 8; ++v) { m8[v] = -3.0e38f; l8[v] = 0.f; }
#pragma unroll
  for (int jd = 0; jd < 4; ++jd) o[jd] = vzero;

  const unsigned short* vbase = Vt + ((size_t)b * H_ + h) * DK_ * S_;

  for (int k0 = 0; k0 < S_; k0 += 64) {
    // --- scores S[16 x 64] = Q * K^T (unscaled), 8 WMMAs ---
    v8f sf[4] = {vzero, vzero, vzero, vzero};
#pragma unroll
    for (int j = 0; j < 4; ++j) {
      const int key = k0 + j * 16 + ln;
      const unsigned short* kp = Kb + ((size_t)key * B_ + b) * D_ + h * DK_;
#pragma unroll
      for (int kq = 0; kq < 2; ++kq) {
        v16bf bk = load_frag_k16(kp + kq * 32 + lh * 16);
        sf[j] = wmma_bf16(aq[kq], bk, sf[j]);
      }
    }

    // --- online softmax: row max / rescale / probabilities ---
    float mn[8], al[8];
#pragma unroll
    for (int v = 0; v < 8; ++v) {
      float x = fmaxf(fmaxf(sf[0][v], sf[1][v]), fmaxf(sf[2][v], sf[3][v]));
      x = fmaxf(x, __shfl_xor(x, 1, 32));
      x = fmaxf(x, __shfl_xor(x, 2, 32));
      x = fmaxf(x, __shfl_xor(x, 4, 32));
      x = fmaxf(x, __shfl_xor(x, 8, 32));   // rows m and m+8 reduce in halves
      mn[v] = fmaxf(m8[v], x);
      al[v] = __expf(m8[v] - mn[v]);
      m8[v] = mn[v];
    }
#pragma unroll
    for (int v = 0; v < 8; ++v) {
      float p[4], rs = 0.f;
#pragma unroll
      for (int j = 0; j < 4; ++j) {
        p[j] = __expf(sf[j][v] - mn[v]);
        rs += p[j];
      }
      rs += __shfl_xor(rs, 1, 32);
      rs += __shfl_xor(rs, 2, 32);
      rs += __shfl_xor(rs, 4, 32);
      rs += __shfl_xor(rs, 8, 32);
      l8[v] = l8[v] * al[v] + rs;
      const int rl = v + (lh << 3);         // local row 0..15
#pragma unroll
      for (int j = 0; j < 4; ++j)
        Pl[rl * PLD + j * 16 + ln] = f2bf(p[j]);  // C-frag -> row-major P
    }
#pragma unroll
    for (int jd = 0; jd < 4; ++jd)
#pragma unroll
      for (int v = 0; v < 8; ++v) o[jd][v] *= al[v];

    // --- O += P(16x64) * V(64x64), V pre-transposed, 8 WMMAs ---
    v16bf ap[2];
#pragma unroll
    for (int i = 0; i < 2; ++i)
      ap[i] = load_frag_a(Pl + ln * PLD + i * 32 + lh * 16);
#pragma unroll
    for (int jd = 0; jd < 4; ++jd) {
      const unsigned short* vp = vbase + (size_t)(jd * 16 + ln) * S_ + k0;
#pragma unroll
      for (int i = 0; i < 2; ++i) {
        v16bf bv = load_frag_k16(vp + i * 32 + lh * 16);
        o[jd] = wmma_bf16(ap[i], bv, o[jd]);
      }
    }
  }

  // --- normalize and store bf16 [S*B, D] ---
#pragma unroll
  for (int jd = 0; jd < 4; ++jd) {
#pragma unroll
    for (int v = 0; v < 8; ++v) {
      const int srow = q0 + v + (lh << 3);
      const int col  = h * DK_ + jd * 16 + ln;
      AO[((size_t)srow * B_ + b) * D_ + col] = f2bf(o[jd][v] / l8[v]);
    }
  }
}

// ---------------------------------------------------------------------------
// Launcher
// ---------------------------------------------------------------------------
extern "C" void kernel_launch(void* const* d_in, const int* in_sizes, int n_in,
                              void* d_out, int out_size, void* d_ws,
                              size_t ws_size, hipStream_t stream) {
  const float* query = (const float*)d_in[0];
  const float* key   = (const float*)d_in[1];
  const float* value = (const float*)d_in[2];
  const float* Wq = (const float*)d_in[3];
  const float* bq = (const float*)d_in[4];
  const float* Wk = (const float*)d_in[5];
  const float* bk = (const float*)d_in[6];
  const float* Wv = (const float*)d_in[7];
  const float* bv = (const float*)d_in[8];
  const float* Wo = (const float*)d_in[9];
  const float* bo = (const float*)d_in[10];

  // Workspace carve-up (bf16 halfwords). Total = 64 MiB.
  const size_t MD = (size_t)M_ * D_;   // 4M elements
  const size_t DD = (size_t)D_ * D_;   // 1M elements
  unsigned short* ws  = (unsigned short*)d_ws;
  unsigned short* Xq  = ws;            // converted activations
  unsigned short* Xk  = Xq + MD;
  unsigned short* Xv  = Xk + MD;
  unsigned short* Qb  = Xv + MD;       // projected Q [S*B, D]
  unsigned short* Kbf = Qb + MD;       // projected K [S*B, D]
  unsigned short* Vt  = Kbf + MD;      // projected V transposed [B,H,DK,S]
  unsigned short* AO  = Vt + MD;       // attention output [S*B, D]
  unsigned short* Wqb = AO + MD;       // bf16 weights
  unsigned short* Wkb = Wqb + DD;
  unsigned short* Wvb = Wkb + DD;
  unsigned short* Wob = Wvb + DD;

  // 1) f32 -> bf16 conversions
  {
    const int n4a = (int)(MD / 4), n4w = (int)(DD / 4);
    const dim3 blk(256);
    cvt_bf16_kernel<<<dim3((n4a + 255) / 256), blk, 0, stream>>>(
        (const float4v*)query, (ushort4v*)Xq, n4a);
    cvt_bf16_kernel<<<dim3((n4a + 255) / 256), blk, 0, stream>>>(
        (const float4v*)key, (ushort4v*)Xk, n4a);
    cvt_bf16_kernel<<<dim3((n4a + 255) / 256), blk, 0, stream>>>(
        (const float4v*)value, (ushort4v*)Xv, n4a);
    cvt_bf16_kernel<<<dim3((n4w + 255) / 256), blk, 0, stream>>>(
        (const float4v*)Wq, (ushort4v*)Wqb, n4w);
    cvt_bf16_kernel<<<dim3((n4w + 255) / 256), blk, 0, stream>>>(
        (const float4v*)Wk, (ushort4v*)Wkb, n4w);
    cvt_bf16_kernel<<<dim3((n4w + 255) / 256), blk, 0, stream>>>(
        (const float4v*)Wv, (ushort4v*)Wvb, n4w);
    cvt_bf16_kernel<<<dim3((n4w + 255) / 256), blk, 0, stream>>>(
        (const float4v*)Wo, (ushort4v*)Wob, n4w);
  }

  // 2) input projections (WMMA GEMMs)
  const dim3 ggrid(D_ / 128, M_ / 128), gblk(256);
  gemm_bf16_kernel<0><<<ggrid, gblk, 0, stream>>>(Xq, Wqb, bq, Qb, M_, D_, D_);
  gemm_bf16_kernel<0><<<ggrid, gblk, 0, stream>>>(Xk, Wkb, bk, Kbf, M_, D_, D_);
  gemm_bf16_kernel<1><<<ggrid, gblk, 0, stream>>>(Xv, Wvb, bv, Vt, M_, D_, D_);

  // 3) flash attention
  flash_attn_kernel<<<dim3(S_ / 64, B_ * H_), dim3(128), 0, stream>>>(Qb, Kbf,
                                                                      Vt, AO);

  // 4) output projection -> f32 d_out
  gemm_bf16_kernel<2><<<ggrid, gblk, 0, stream>>>(AO, Wob, bo, (float*)d_out,
                                                  M_, D_, D_);
}
